// SelfAttention_41360535061076
// MI455X (gfx1250) — matrix operand order repaired and checked by reading, not compiled, not run on hardware
//
#include <hip/hip_runtime.h>
#include <hip/hip_bf16.h>

typedef _Float16 v16h __attribute__((ext_vector_type(16)));
typedef _Float16 v8h  __attribute__((ext_vector_type(8)));
typedef float    v8f  __attribute__((ext_vector_type(8)));
typedef int      v4i  __attribute__((vector_size(16)));   // matches builtin param

#define B_   4
#define C_   256
#define N_   4096
#define CQK  32

#define GAS __attribute__((address_space(1)))
#define LAS __attribute__((address_space(3)))

#ifndef __has_builtin
#define __has_builtin(x) 0
#endif

#if __has_builtin(__builtin_amdgcn_global_load_async_to_lds_b128)
#define HAVE_ASYNC 1
#else
#define HAVE_ASYNC 0
#endif

#if HAVE_ASYNC
__device__ __forceinline__ void async_cp16(const _Float16* gp, _Float16* lp) {
  __builtin_amdgcn_global_load_async_to_lds_b128((GAS v4i*)gp, (LAS v4i*)lp, 0, 0);
}
__device__ __forceinline__ void wait_async_le9() {
#if __has_builtin(__builtin_amdgcn_s_wait_asynccnt)
  __builtin_amdgcn_s_wait_asynccnt(9);
#else
  asm volatile("s_wait_asynccnt 0x9" ::: "memory");
#endif
}
__device__ __forceinline__ void wait_async_le0() {
#if __has_builtin(__builtin_amdgcn_s_wait_asynccnt)
  __builtin_amdgcn_s_wait_asynccnt(0);
#else
  asm volatile("s_wait_asynccnt 0x0" ::: "memory");
#endif
}
#endif

// ---------------------------------------------------------------------------
// Fragment helpers: contiguous k order (k = 16*half + s), same for A and B.
// ---------------------------------------------------------------------------

// B-matrix 32x16 (KxN) from a row-major [N_rows x rowStride] array whose
// contiguous dim is K: lane L<16 -> column L, K=0..15; lane L+16 -> K=16..31.
__device__ __forceinline__ v16h load_b_frag_g(const _Float16* __restrict__ rowbase,
                                              int rowStride, int lane) {
  int r = lane & 15;
  const _Float16* p = rowbase + (size_t)r * rowStride + ((lane >> 4) << 4);
  return *(const v16h*)p;
}

// Assemble a 16-half fragment chunk pair from a (16B-aligned) LDS row pointer.
__device__ __forceinline__ v16h frag_from_row16(const _Float16* p) {
  v8h a = *(const v8h*)p;
  v8h b = *(const v8h*)(p + 8);
  v16h r;
#pragma unroll
  for (int i = 0; i < 8; ++i) { r[i] = a[i]; r[8 + i] = b[i]; }
  return r;
}

// A-matrix 16x32 (MxK): lane L<16 holds row L, K=0..15;
// lane L+16 holds row L, K=16..31.
__device__ __forceinline__ v16h load_a_frag_g(const _Float16* __restrict__ rowbase,
                                              int rowStride, int lane) {
  int half = lane >> 4;
  int r = lane & 15;
  const _Float16* p = rowbase + (size_t)r * rowStride + half * 16;
  v8h c0 = *(const v8h*)p;
  v8h c1 = *(const v8h*)(p + 8);
  v16h out;
#pragma unroll
  for (int i = 0; i < 8; ++i) { out[i] = c0[i]; out[8 + i] = c1[i]; }
  return out;
}

// ---------------------------------------------------------------------------
// Kernel 1: fused 1x1-conv projections.  grid = (N/128, 5, B), block = 128.
// q,k stored [B][N][32] f16; v stored [B][C][N] f16 (B-fragment friendly).
// ---------------------------------------------------------------------------
__global__ __launch_bounds__(128) void proj_kernel(
    const float* __restrict__ x,
    const float* __restrict__ wq, const float* __restrict__ bq,
    const float* __restrict__ wk, const float* __restrict__ bk,
    const float* __restrict__ wv, const float* __restrict__ bv,
    _Float16* __restrict__ qh, _Float16* __restrict__ kh,
    _Float16* __restrict__ vh) {
  __shared__ float wl[64 * 256];
  const int tid = threadIdx.x;
  const int g = blockIdx.y;
  const int b = blockIdx.z;
  const int n = blockIdx.x * 128 + tid;

  for (int e = tid; e < 64 * 256; e += 128) {
    float w;
    if (g == 0) w = (e < 32 * 256) ? wq[e] : wk[e - 32 * 256];
    else        w = wv[(size_t)(g - 1) * 64 * 256 + e];
    wl[e] = w;
  }
  __syncthreads();

  float acc[64];
#pragma unroll
  for (int t = 0; t < 64; ++t) {
    if (g == 0) acc[t] = (t < 32) ? bq[t] : bk[t - 32];
    else        acc[t] = bv[(g - 1) * 64 + t];
  }

  const float* xp = x + (size_t)b * C_ * N_ + n;
  for (int c = 0; c < C_; ++c) {
    float xv = xp[(size_t)c * N_];
#pragma unroll
    for (int t = 0; t < 64; ++t) acc[t] += wl[t * 256 + c] * xv;
  }

  if (g == 0) {
    _Float16* qp = qh + ((size_t)b * N_ + n) * CQK;
    _Float16* kp = kh + ((size_t)b * N_ + n) * CQK;
#pragma unroll
    for (int t = 0; t < 32; ++t) qp[t] = (_Float16)acc[t];
#pragma unroll
    for (int t = 0; t < 32; ++t) kp[t] = (_Float16)acc[32 + t];
  } else {
    _Float16* vp = vh + ((size_t)b * C_ + (size_t)(g - 1) * 64) * N_ + n;
#pragma unroll
    for (int t = 0; t < 64; ++t) vp[(size_t)t * N_] = (_Float16)acc[t];
  }
}

// ---------------------------------------------------------------------------
// Kernel 2: row stats.  m = max_j S, s = sum_j exp(S)  (bounded logits: the
// max subtraction is deferred to pass 2 as exp(m)/s, halving exp count here).
// One wave per 16-row i-tile; one WMMA per 16-wide j-tile (K dim == 32).
// ---------------------------------------------------------------------------
__global__ __launch_bounds__(128) void stats_kernel(
    const _Float16* __restrict__ qh, const _Float16* __restrict__ kh,
    float* __restrict__ mrow, float* __restrict__ lrow) {
  const int lane = threadIdx.x & 31;
  const int wid = threadIdx.x >> 5;
  const int tiles_per_b = N_ / 64;
  const int b = blockIdx.x / tiles_per_b;
  const int ibase = (blockIdx.x % tiles_per_b) * 64 + wid * 16;

  const _Float16* qb = qh + ((size_t)b * N_ + ibase) * CQK;
  const _Float16* kb = kh + (size_t)b * N_ * CQK;

  v16h aq = load_a_frag_g(qb, CQK, lane);

  float m[8], s[8];
#pragma unroll
  for (int r = 0; r < 8; ++r) { m[r] = -1e30f; s[r] = 0.0f; }

  for (int j = 0; j < N_; j += 16) {
    v16h bk = load_b_frag_g(kb + (size_t)j * CQK, CQK, lane);
    v8f sv = {};
    sv = __builtin_amdgcn_wmma_f32_16x16x32_f16(false, aq, false, bk,
                                                (short)0, sv, false, false);
#pragma unroll
    for (int r = 0; r < 8; ++r) {
      float xv = sv[r];
      m[r] = fmaxf(m[r], xv);
      s[r] += __expf(xv);
    }
  }

  // Reduce across the 16 lanes holding one row (xor masks stay in-half).
#pragma unroll
  for (int off = 1; off < 16; off <<= 1) {
#pragma unroll
    for (int r = 0; r < 8; ++r) {
      m[r] = fmaxf(m[r], __shfl_xor(m[r], off));
      s[r] += __shfl_xor(s[r], off);
    }
  }

  if ((lane & 15) == 0) {
    int half = lane >> 4;
#pragma unroll
    for (int r = 0; r < 8; ++r) {
      int row = ibase + half * 8 + r;
      mrow[(size_t)b * N_ + row] = m[r];
      lrow[(size_t)b * N_ + row] = s[r];
    }
  }
}

// ---------------------------------------------------------------------------
// Kernel 3: O = softmax(QK^T) V.  Block = 4 waves sharing one j-stream:
// K (32x32) and V (256x32) j-chunks are staged once per block into
// double-buffered LDS via async global->LDS DMA (ASYNCcnt-tracked), removing
// the 4x redundant per-wave L2 traffic.  P goes D-layout -> LDS -> A-layout
// inside each wave (DS ops are wave-in-order; compiler emits s_wait_dscnt).
// ---------------------------------------------------------------------------
__global__ __launch_bounds__(128, 1) void attn_kernel(
    const _Float16* __restrict__ qh, const _Float16* __restrict__ kh,
    const _Float16* __restrict__ vh, const float* __restrict__ mrow,
    const float* __restrict__ lrow, const float* __restrict__ x,
    const float* __restrict__ gamma, float* __restrict__ out) {
  // 80B row stride: starting banks distinct mod 64 -> conflict-free 16B reads.
  __shared__ __align__(16) _Float16 kt[2][32][40];
  __shared__ __align__(16) _Float16 vt[2][256][40];
  __shared__ __align__(16) _Float16 plds[4][16][40];

  const int tid = threadIdx.x;
  const int lane = tid & 31;
  const int wid = tid >> 5;
  const int half = lane >> 4;
  const int col = lane & 15;
  const int tiles_per_b = N_ / 64;
  const int b = blockIdx.x / tiles_per_b;
  const int ibase = (blockIdx.x % tiles_per_b) * 64 + wid * 16;

  const _Float16* qb = qh + ((size_t)b * N_ + ibase) * CQK;
  const _Float16* kb = kh + (size_t)b * N_ * CQK;
  const _Float16* vb = vh + (size_t)b * C_ * N_;

  v16h aq = load_a_frag_g(qb, CQK, lane);

  float mr[8], li[8];
#pragma unroll
  for (int r = 0; r < 8; ++r) {
    int row = ibase + half * 8 + r;
    float mm = mrow[(size_t)b * N_ + row];
    mr[r] = mm;
    li[r] = __expf(mm) / lrow[(size_t)b * N_ + row];  // P = exp(S-m)*exp(m)/s
  }

  v8f acc[16];
  v8f zero = {};
#pragma unroll
  for (int t = 0; t < 16; ++t) acc[t] = zero;

  _Float16 (*pt)[40] = plds[wid];

  // --- stage one 32-wide j-chunk of K and V into buffer bi (9 x b128/thread)
  auto stage = [&](int bi, int j) {
    {
      int row = tid >> 2, sub = tid & 3;                 // 32 rows x 4 chunks
      const _Float16* gp = kb + (size_t)(j + row) * CQK + sub * 8;
      _Float16* lp = &kt[bi][row][sub * 8];
#if HAVE_ASYNC
      async_cp16(gp, lp);
#else
      *(v8h*)lp = *(const v8h*)gp;
#endif
    }
#pragma unroll
    for (int s = 0; s < 8; ++s) {                        // 256 rows x 4 chunks
      int chunk = s * 128 + tid;
      int row = chunk >> 2, sub = chunk & 3;
      const _Float16* gp = vb + (size_t)row * N_ + j + sub * 8;
      _Float16* lp = &vt[bi][row][sub * 8];
#if HAVE_ASYNC
      async_cp16(gp, lp);
#else
      *(v8h*)lp = *(const v8h*)gp;
#endif
    }
  };

  const int NT = N_ / 32;
  stage(0, 0);

  for (int jt = 0; jt < NT; ++jt) {
    const int cur = jt & 1;
    if (jt + 1 < NT) {
      stage(cur ^ 1, (jt + 1) * 32);
#if HAVE_ASYNC
      wait_async_le9();   // current chunk's 9 loads done, next stays in flight
#endif
    } else {
#if HAVE_ASYNC
      wait_async_le0();
#endif
    }
    __syncthreads();

    // S = Q K^T for this 32-wide j-chunk (two 16-col WMMAs, K dim = 32).
    v16h bk0 = frag_from_row16(&kt[cur][col][half * 16]);
    v16h bk1 = frag_from_row16(&kt[cur][16 + col][half * 16]);
    v8f s0 = {}, s1 = {};
    s0 = __builtin_amdgcn_wmma_f32_16x16x32_f16(false, aq, false, bk0,
                                                (short)0, s0, false, false);
    s1 = __builtin_amdgcn_wmma_f32_16x16x32_f16(false, aq, false, bk1,
                                                (short)0, s1, false, false);

    // P = exp(S - m) * (exp(m)/l), D-layout -> per-wave LDS tile.
#pragma unroll
    for (int r = 0; r < 8; ++r) {
      float p0 = __expf(s0[r] - mr[r]) * li[r];
      float p1 = __expf(s1[r] - mr[r]) * li[r];
      pt[half * 8 + r][col]      = (_Float16)p0;
      pt[half * 8 + r][col + 16] = (_Float16)p1;
    }
    // Re-read as A-layout fragment (16x32, K = j-within-chunk).
    v8h c0 = *(const v8h*)&pt[col][half * 16];
    v8h c1 = *(const v8h*)&pt[col][half * 16 + 8];
    v16h pa;
#pragma unroll
    for (int i = 0; i < 8; ++i) { pa[i] = c0[i]; pa[8 + i] = c1[i]; }

    // O += P V over all 16 channel tiles, V fragments from shared LDS tile.
#pragma unroll
    for (int t = 0; t < 16; ++t) {
      v16h bv = frag_from_row16(&vt[cur][t * 16 + col][half * 16]);
      acc[t] = __builtin_amdgcn_wmma_f32_16x16x32_f16(false, pa, false, bv,
                                                      (short)0, acc[t], false, false);
    }
    __syncthreads();
  }

  const float g = *gamma;
#pragma unroll
  for (int t = 0; t < 16; ++t) {
#pragma unroll
    for (int r = 0; r < 8; ++r) {
      int i = ibase + half * 8 + r;      // D-layout: VGPR r -> M = r (+8 hi half)
      int c = t * 16 + col;              //           lane   -> N = c
      size_t idx = ((size_t)b * C_ + c) * N_ + i;
      out[idx] = g * acc[t][r] + x[idx];
    }
  }
}

// ---------------------------------------------------------------------------
// Launch.  Workspace: qh/kh [B][N][32] f16 (1 MB each), vh [B][C][N] f16 (8 MB),
// mrow/lrow [B][N] f32 (64 KB each)  ->  ~10.2 MB.
// ---------------------------------------------------------------------------
extern "C" void kernel_launch(void* const* d_in, const int* in_sizes, int n_in,
                              void* d_out, int out_size, void* d_ws, size_t ws_size,
                              hipStream_t stream) {
  const float* input = (const float*)d_in[0];
  const float* wq    = (const float*)d_in[1];
  const float* bq    = (const float*)d_in[2];
  const float* wk    = (const float*)d_in[3];
  const float* bk    = (const float*)d_in[4];
  const float* wv    = (const float*)d_in[5];
  const float* bv    = (const float*)d_in[6];
  const float* gamma = (const float*)d_in[7];
  float* out = (float*)d_out;

  _Float16* qh = (_Float16*)d_ws;
  _Float16* kh = qh + (size_t)B_ * N_ * CQK;
  _Float16* vh = kh + (size_t)B_ * N_ * CQK;
  float* mrow = (float*)(vh + (size_t)B_ * C_ * N_);
  float* lrow = mrow + (size_t)B_ * N_;

  proj_kernel<<<dim3(N_ / 128, 5, B_), 128, 0, stream>>>(
      input, wq, bq, wk, bk, wv, bv, qh, kh, vh);
  stats_kernel<<<dim3(B_ * N_ / 64), 128, 0, stream>>>(qh, kh, mrow, lrow);
  attn_kernel<<<dim3(B_ * N_ / 64), 128, 0, stream>>>(
      qh, kh, vh, mrow, lrow, input, gamma, out);
}
